// RNN_LSTM_twin_57028575756766
// MI455X (gfx1250) — matrix operand order repaired and checked by reading, not compiled, not run on hardware
//
#include <hip/hip_runtime.h>
#include <cstdint>
#include <cstddef>

// ---------------------------------------------------------------------------
// CDNA5 (gfx1250, wave32) WMMA types: v_wmma_f32_16x16x32_bf16
// ---------------------------------------------------------------------------
typedef __bf16 bf16_t;
typedef __attribute__((ext_vector_type(16))) __bf16 v16bf;
typedef __attribute__((ext_vector_type(8)))  __bf16 v8bf;
typedef __attribute__((ext_vector_type(8)))  float  v8f;

namespace {
constexpr int kB  = 32;
constexpr int kT  = 1024;
constexpr int kI  = 512;
constexpr int kH  = 1024;
constexpr int kC  = 512;
constexpr int kG  = 4 * kH;   // 4096 gates (i,f,g,o)
constexpr int kBT = kB * kT;  // 32768
constexpr int kNB = 16;       // persistent-recurrence blocks (co-resident)
}

// A fragment (16-bit A 16x32): p points at A[m][k0 + half*16];
// v = 16 contiguous bf16 (two b128 loads).
__device__ __forceinline__ v16bf ldfragA(const bf16_t* p) {
  union { v16bf v; v8bf h[2]; } u;
  u.h[0] = *(const v8bf*)(p);
  u.h[1] = *(const v8bf*)(p + 8);
  return u.v;
}
// B fragment (16-bit B 32x16): p points at W[n][k0 + half*16];
// v = 16 contiguous bf16 (two b128 loads).
__device__ __forceinline__ v16bf ldfragB(const bf16_t* p) {
  union { v16bf v; v8bf h[2]; } u;
  u.h[0] = *(const v8bf*)(p);
  u.h[1] = *(const v8bf*)(p + 8);
  return u.v;
}

__device__ __forceinline__ float sigmoidf_(float x) {
  return 1.f / (1.f + __expf(-x));
}

// ---------------------------------------------------------------------------
// one-time f32 -> bf16 conversion
// ---------------------------------------------------------------------------
__global__ void cvt_f32_bf16(const float* __restrict__ in,
                             bf16_t* __restrict__ out, int n) {
  int i = blockIdx.x * blockDim.x + threadIdx.x;
  if (i < n) out[i] = (bf16_t)in[i];
}

// zero h ping-buffer 0 and the grid-barrier counter (every launch, so graph
// replays are deterministic)
__global__ void init_state(bf16_t* __restrict__ hping, unsigned* __restrict__ cnt) {
  int i = blockIdx.x * blockDim.x + threadIdx.x;
  if (i < kB * kH) hping[i] = (bf16_t)0.f;
  if (i == 0) *cnt = 0u;
}

// ---------------------------------------------------------------------------
// Big WMMA GEMM: C[M,N] = A[M,K](bf16) * W[N,K](bf16)^T + bias.
// One wave computes a (16*MT) x (16*NT) block.
// K-loop: explicit 2-deep software pipeline with NAMED ping/pong fragment
// buffers inside a `#pragma unroll 1` loop. Loads for k-step i+1 are issued
// before the WMMAs consuming k-step i, so waits become s_wait_loadcnt<=N
// (overlap) while VGPR pressure stays bounded (~190) -> no scratch spills,
// no LICM hoist-and-spill.
// ---------------------------------------------------------------------------
template<int MT, int NT, int K>
__global__ __launch_bounds__(256) void gemm_bf16_wmma(
    const bf16_t* __restrict__ A, int lda,
    const bf16_t* __restrict__ W, int ldw,     // [N,K] row-major
    float* __restrict__ Cd, int ldc,
    const float* __restrict__ bias,            // [N]
    int M, int N) {
  const int wave = (int)((blockIdx.x * 256u + threadIdx.x) >> 5);
  const int wpr  = N / (16 * NT);
  const int m0   = (wave / wpr) * (16 * MT);
  const int n0   = (wave % wpr) * (16 * NT);
  if (m0 >= M) return;  // uniform per wave -> EXEC stays all-ones for WMMA

  const int l = threadIdx.x & 31, l15 = l & 15, lh = l >> 4;

  const bf16_t* arow[MT];
#pragma unroll
  for (int mi = 0; mi < MT; ++mi)
    arow[mi] = A + (size_t)(m0 + 16 * mi + l15) * lda + lh * 16;
  const bf16_t* brow[NT];
#pragma unroll
  for (int t = 0; t < NT; ++t)
    brow[t] = W + (size_t)(n0 + 16 * t + l15) * ldw + lh * 16;

  v8f acc[MT][NT] = {};
  v16bf a0[MT], a1[MT], b0[NT], b1[NT];
  constexpr int KS = K / 32;  // 16 or 32 (even)

#pragma unroll
  for (int mi = 0; mi < MT; ++mi) a0[mi] = ldfragA(arow[mi]);
#pragma unroll
  for (int t = 0; t < NT; ++t) b0[t] = ldfragB(brow[t]);

#pragma unroll 1
  for (int ks = 0; ks < KS - 2; ks += 2) {
    const int k1 = (ks + 1) * 32, k2 = (ks + 2) * 32;
#pragma unroll
    for (int mi = 0; mi < MT; ++mi) a1[mi] = ldfragA(arow[mi] + k1);
#pragma unroll
    for (int t = 0; t < NT; ++t) b1[t] = ldfragB(brow[t] + k1);
#pragma unroll
    for (int mi = 0; mi < MT; ++mi)
#pragma unroll
      for (int t = 0; t < NT; ++t)
        acc[mi][t] = __builtin_amdgcn_wmma_f32_16x16x32_bf16(
            false, a0[mi], false, b0[t], (short)0, acc[mi][t], false, false);
#pragma unroll
    for (int mi = 0; mi < MT; ++mi) a0[mi] = ldfragA(arow[mi] + k2);
#pragma unroll
    for (int t = 0; t < NT; ++t) b0[t] = ldfragB(brow[t] + k2);
#pragma unroll
    for (int mi = 0; mi < MT; ++mi)
#pragma unroll
      for (int t = 0; t < NT; ++t)
        acc[mi][t] = __builtin_amdgcn_wmma_f32_16x16x32_bf16(
            false, a1[mi], false, b1[t], (short)0, acc[mi][t], false, false);
  }
  {  // tail: k-steps KS-2 (in buf0) and KS-1
    const int k1 = (KS - 1) * 32;
#pragma unroll
    for (int mi = 0; mi < MT; ++mi) a1[mi] = ldfragA(arow[mi] + k1);
#pragma unroll
    for (int t = 0; t < NT; ++t) b1[t] = ldfragB(brow[t] + k1);
#pragma unroll
    for (int mi = 0; mi < MT; ++mi)
#pragma unroll
      for (int t = 0; t < NT; ++t)
        acc[mi][t] = __builtin_amdgcn_wmma_f32_16x16x32_bf16(
            false, a0[mi], false, b0[t], (short)0, acc[mi][t], false, false);
#pragma unroll
    for (int mi = 0; mi < MT; ++mi)
#pragma unroll
      for (int t = 0; t < NT; ++t)
        acc[mi][t] = __builtin_amdgcn_wmma_f32_16x16x32_bf16(
            false, a1[mi], false, b1[t], (short)0, acc[mi][t], false, false);
  }

#pragma unroll
  for (int t = 0; t < NT; ++t) {
    const int n = n0 + 16 * t + l15;
    const float bv = bias[n];
#pragma unroll
    for (int mi = 0; mi < MT; ++mi) {
#pragma unroll
      for (int r = 0; r < 8; ++r) {
        const int m = m0 + 16 * mi + lh * 8 + r;
        Cd[(size_t)m * ldc + n] = acc[mi][t][r] + bv;
      }
    }
  }
}

// ---------------------------------------------------------------------------
// Persistent fused LSTM recurrence: ONE launch for all 1024 timesteps.
// Wave = (m0 in {0,16}, j0): its 4 accumulator tiles sit at gate columns
// {j0, j0+H, j0+2H, j0+3H} so i/f/g/o land on the SAME 16x16 (b,j) patch and
// the cell update runs entirely in registers; c stays in VGPRs for all steps.
// W_hh streams from L2 each step (8MB bf16, L2-resident). xg gate inputs
// (HBM, 512MB streamed once) are prefetched BEFORE the K-loop so their
// latency hides behind the 128 WMMAs. Same 2-deep pipelined K-loop as above.
// ---------------------------------------------------------------------------
__global__ __launch_bounds__(256) void lstm_recurrence(
    const bf16_t* __restrict__ Whh,   // [4H,H] bf16
    const float*  __restrict__ b_hh,  // [4H]
    const float*  __restrict__ xg,    // [B,T,4H] = x@W_ih^T + b_ih
    bf16_t* __restrict__ hping,       // [2][B*H]
    bf16_t* __restrict__ hsBT,        // [B,T,H]
    bf16_t* __restrict__ hsTB,        // [T,B,H]
    unsigned* __restrict__ bar) {
  const int wave = (int)((blockIdx.x * 256u + threadIdx.x) >> 5);  // 0..127
  const int m0 = (wave >> 6) * 16;   // 0 or 16
  const int j0 = (wave & 63) * 16;   // hidden-column group
  const int l = threadIdx.x & 31, l15 = l & 15, lh = l >> 4;

  const bf16_t* brow[4];
  float bh[4];
#pragma unroll
  for (int t = 0; t < 4; ++t) {
    const int n = t * kH + j0 + l15;
    brow[t] = Whh + (size_t)n * kH + lh * 16;
    bh[t] = b_hh[n];
  }
  const int j = j0 + l15;
  const int mbase = m0 + lh * 8;
  const float* xrow = xg + (size_t)mbase * kT * kG + j;  // + r*T*4H + s*4H + t*H

  float creg[8] = {};
  constexpr int KS = kH / 32;  // 32

#pragma unroll 1
  for (int s = 0; s < kT; ++s) {
    const bf16_t* hin  = hping + (size_t)(s & 1) * (kB * kH);
    const bf16_t* arow = hin + (size_t)(m0 + l15) * kH + lh * 16;

    // prefetch this step's gate inputs (HBM) so they overlap the K-loop
    float xv[4][8];
    {
      const float* xps = xrow + (size_t)s * kG;
#pragma unroll
      for (int r = 0; r < 8; ++r)
#pragma unroll
        for (int t = 0; t < 4; ++t)
          xv[t][r] = xps[(size_t)r * kT * kG + t * kH];
    }

    v8f acc[4] = {};
    v16bf a0, a1, b0[4], b1[4];
    a0 = ldfragA(arow);
#pragma unroll
    for (int t = 0; t < 4; ++t) b0[t] = ldfragB(brow[t]);

#pragma unroll 1
    for (int ks = 0; ks < KS - 2; ks += 2) {
      const int k1 = (ks + 1) * 32, k2 = (ks + 2) * 32;
      a1 = ldfragA(arow + k1);
#pragma unroll
      for (int t = 0; t < 4; ++t) b1[t] = ldfragB(brow[t] + k1);
#pragma unroll
      for (int t = 0; t < 4; ++t)
        acc[t] = __builtin_amdgcn_wmma_f32_16x16x32_bf16(
            false, a0, false, b0[t], (short)0, acc[t], false, false);
      a0 = ldfragA(arow + k2);
#pragma unroll
      for (int t = 0; t < 4; ++t) b0[t] = ldfragB(brow[t] + k2);
#pragma unroll
      for (int t = 0; t < 4; ++t)
        acc[t] = __builtin_amdgcn_wmma_f32_16x16x32_bf16(
            false, a1, false, b1[t], (short)0, acc[t], false, false);
    }
    {
      const int k1 = (KS - 1) * 32;
      a1 = ldfragA(arow + k1);
#pragma unroll
      for (int t = 0; t < 4; ++t) b1[t] = ldfragB(brow[t] + k1);
#pragma unroll
      for (int t = 0; t < 4; ++t)
        acc[t] = __builtin_amdgcn_wmma_f32_16x16x32_bf16(
            false, a0, false, b0[t], (short)0, acc[t], false, false);
#pragma unroll
      for (int t = 0; t < 4; ++t)
        acc[t] = __builtin_amdgcn_wmma_f32_16x16x32_bf16(
            false, a1, false, b1[t], (short)0, acc[t], false, false);
    }

    // in-register LSTM cell update + h emission
    bf16_t* hout = hping + (size_t)((s + 1) & 1) * (kB * kH);
#pragma unroll
    for (int r = 0; r < 8; ++r) {
      const int m = mbase + r;
      const float gi = acc[0][r] + bh[0] + xv[0][r];
      const float gf = acc[1][r] + bh[1] + xv[1][r];
      const float gg = acc[2][r] + bh[2] + xv[2][r];
      const float go = acc[3][r] + bh[3] + xv[3][r];
      const float cc = sigmoidf_(gf) * creg[r] + sigmoidf_(gi) * tanhf(gg);
      const float hh = sigmoidf_(go) * tanhf(cc);
      creg[r] = cc;
      const bf16_t hb = (bf16_t)hh;
      hout[(size_t)m * kH + j] = hb;                    // next-step GEMM operand
      hsBT[((size_t)m * kT + s) * kH + j] = hb;         // [B,T,H] for fc GEMM
      hsTB[((size_t)s * kB + m) * kH + j] = hb;         // [T,B,H] for ln GEMM
    }

    // ---- device-wide step barrier (release/acquire, agent scope) ----
    __threadfence();
    __syncthreads();
    if (threadIdx.x == 0) {
      __hip_atomic_fetch_add(bar, 1u, __ATOMIC_RELEASE, __HIP_MEMORY_SCOPE_AGENT);
      const unsigned target = (unsigned)(s + 1) * (unsigned)kNB;
      while (__hip_atomic_load(bar, __ATOMIC_ACQUIRE, __HIP_MEMORY_SCOPE_AGENT) < target)
        __builtin_amdgcn_s_sleep(2);
    }
    __syncthreads();
  }
}

// ---------------------------------------------------------------------------
// kernel_launch: 10 total launches
// ---------------------------------------------------------------------------
extern "C" void kernel_launch(void* const* d_in, const int* in_sizes, int n_in,
                              void* d_out, int out_size, void* d_ws, size_t ws_size,
                              hipStream_t stream) {
  (void)in_sizes; (void)n_in; (void)out_size; (void)ws_size;

  const float* x    = (const float*)d_in[0];
  const float* Wih  = (const float*)d_in[1];
  const float* Whh  = (const float*)d_in[2];
  const float* b_ih = (const float*)d_in[3];
  const float* b_hh = (const float*)d_in[4];
  const float* fcw  = (const float*)d_in[5];
  const float* fcb  = (const float*)d_in[6];
  const float* lnw  = (const float*)d_in[7];
  const float* lnb  = (const float*)d_in[8];

  float* out    = (float*)d_out;               // [B,T,C]
  float* states = out + (size_t)kBT * kC;      // [T,B,H]

  char* w = (char*)d_ws;
  auto carve = [&](size_t bytes) -> void* {
    void* p = (void*)w;
    w += (bytes + 255) & ~(size_t)255;
    return p;
  };
  bf16_t*   x_bf   = (bf16_t*)carve((size_t)kBT * kI * 2);
  bf16_t*   Wih_bf = (bf16_t*)carve((size_t)kG * kI * 2);
  bf16_t*   Whh_bf = (bf16_t*)carve((size_t)kG * kH * 2);
  bf16_t*   fcw_bf = (bf16_t*)carve((size_t)kC * kH * 2);
  bf16_t*   lnw_bf = (bf16_t*)carve((size_t)kH * kH * 2);
  float*    xg     = (float*)carve((size_t)kBT * kG * 4);   // 512 MB
  bf16_t*   hping  = (bf16_t*)carve((size_t)2 * kB * kH * 2);
  bf16_t*   hsBT   = (bf16_t*)carve((size_t)kBT * kH * 2);
  bf16_t*   hsTB   = (bf16_t*)carve((size_t)kBT * kH * 2);
  unsigned* bar    = (unsigned*)carve(256);

  auto cvt = [&](const float* src, bf16_t* dst, int n) {
    cvt_f32_bf16<<<(n + 255) / 256, 256, 0, stream>>>(src, dst, n);
  };
  cvt(x,   x_bf,   kBT * kI);
  cvt(Wih, Wih_bf, kG * kI);
  cvt(Whh, Whh_bf, kG * kH);
  cvt(fcw, fcw_bf, kC * kH);
  cvt(lnw, lnw_bf, kH * kH);

  init_state<<<(kB * kH) / 256, 256, 0, stream>>>(hping, bar);

  // phase 1: xg = x @ W_ih^T + b_ih   (M=32768, N=4096, K=512)
  {
    const int M = kBT, N = kG;
    const int waves = (M / 32) * (N / 64);
    gemm_bf16_wmma<2, 4, kI><<<waves / 8, 256, 0, stream>>>(
        x_bf, kI, Wih_bf, kI, xg, kG, b_ih, M, N);
  }

  // phase 2: full recurrence in one persistent launch
  lstm_recurrence<<<kNB, 256, 0, stream>>>(Whh_bf, b_hh, xg, hping, hsBT, hsTB, bar);

  // phase 3: out = hs @ fc_w^T + fc_b   (M=32768, N=512, K=1024)
  {
    const int M = kBT, N = kC;
    const int waves = (M / 32) * (N / 64);
    gemm_bf16_wmma<2, 4, kH><<<waves / 8, 256, 0, stream>>>(
        hsBT, kH, fcw_bf, kH, out, kC, fcb, M, N);
  }

  // phase 4: states = hs[T,B,H] @ ln_w^T + ln_b   (M=32768, N=1024, K=1024)
  {
    const int M = kBT, N = kH;
    const int waves = (M / 32) * (N / 64);
    gemm_bf16_wmma<2, 4, kH><<<waves / 8, 256, 0, stream>>>(
        hsTB, kH, lnw_bf, kH, states, kH, lnb, M, N);
  }
}
